// FlexChannelPConv2D_25821343383780
// MI455X (gfx1250) — matrix operand order repaired and checked by reading, not compiled, not run on hardware
//
#include <hip/hip_runtime.h>

// ---------------------------------------------------------------------------
// FlexChannelPConv2D fused kernel for MI455X (gfx1250, wave32, WMMA).
//
// pp[b,q,c,f]  = sum_p data[b,p,c] * filt[p,q,f]            (f16 WMMA, f32 acc)
// out[b,q,co]  = sum_fsel premul(pp)[b,q,fsel] * W[fsel,co] (complex, VALU)
//
// Filter synthesis avoids atan2/sincos entirely: cos(n*theta)/sin(n*theta)
// come from dx/r, dy/r and Chebyshev-style angle recurrences (frequencies are
// small integers by construction).
//
// Grid: 100 blocks (one per 16-wide q tile). Block: 288 threads = 9 waves,
// wave w owns basis function f = w with 4 f32 accumulators (2 bc-tiles x re/im).
// ---------------------------------------------------------------------------

typedef __attribute__((ext_vector_type(16))) _Float16 v16h;
typedef __attribute__((ext_vector_type(8)))  float    v8f;

#define PQ    1600   // H*W
#define CIN   8
#define NB    4
#define NF    9      // basis functions
#define FSEL  48     // kept (channel,basis) pairs
#define COUT  36
#define KT    32     // K (p) tile per WMMA step
#define QT    16     // q tile (WMMA N)
#define NTHR  288    // 9 waves of 32

__global__ __launch_bounds__(NTHR) void flexpconv_fused(
    const float* __restrict__ data,        // [NB, PQ, CIN]
    const float* __restrict__ grid,        // [2, PQ]
    const float* __restrict__ weights,     // [nnz, 2]
    const float* __restrict__ b_rad,       // [NF]
    const float* __restrict__ b_sig,       // [NF]
    const float* __restrict__ b_frq,       // [NF]
    const float* __restrict__ pre_mul,     // [FSEL, 2]
    const int*   __restrict__ pre_cut_idx, // [FSEL]
    const int*   __restrict__ conn_rows,   // [nnz]
    const int*   __restrict__ conn_cols,   // [nnz]
    const int*   __restrict__ out_rot,     // [COUT]
    float*       __restrict__ out,         // [NB, PQ, COUT, 2]
    int nnz)
{
    // Fragment buffers (live during K loop) overlaid with the dense second-stage
    // weight matrix (live only after the K loop) to stay well under LDS limits.
    __shared__ union {
        struct {
            __align__(32) _Float16 B[NF][2][32][16]; // B fragments: [f][re/im][lane][16 halfs]
            __align__(32) _Float16 A[2][32][16];     // A fragments: [bc-tile][lane][16 halfs]
        } s1;
        struct {
            float W[FSEL][COUT][2];                  // dense wfull
            float Mul[FSEL][2];
            int   Idx[FSEL];
        } s2;
    } u;
    __shared__ float sPP[NB][QT][CIN][NF][2];        // 36 KB pp tile
    __shared__ float sQy[QT], sQx[QT];

    const int t    = threadIdx.x;
    const int lane = t & 31;
    const int w    = t >> 5;                 // wave id == basis index f
    const int q0   = blockIdx.x * QT;

    // Uniform basis constants -> registers (scalar loads).
    float br[NF], binv[NF], fsgn[NF];
    int   anf[NF];                           // |frequency| as small integer
#pragma unroll
    for (int f = 0; f < NF; ++f) {
        br[f]   = b_rad[f];
        binv[f] = 1.0f / (1.41421356237f * b_sig[f]);
        float fq = b_frq[f];
        int   n  = __float2int_rn(fq);
        anf[f]   = (n < 0) ? -n : n;
        fsgn[f]  = (n < 0) ? -1.0f : 1.0f;
    }

    if (t < QT) { sQy[t] = grid[q0 + t]; sQx[t] = grid[PQ + q0 + t]; }

    v8f acc00 = {}, acc01 = {}, acc10 = {}, acc11 = {}; // [bc-tile][re/im]

    for (int k0 = 0; k0 < PQ; k0 += KT) {
        __syncthreads();   // previous iteration's fragment reads done

        // ---- stage data tile into A-fragment layout (ISA 16-bit A 16x32) ----
        for (int i = t; i < 32 * KT; i += NTHR) {
            int bc = i >> 5;                 // 0..31  (bc = b*8 + c)
            int pl = i & 31;                 // local K (p) index
            float v = data[((bc >> 3) * PQ + (k0 + pl)) * CIN + (bc & 7)];
            int m = bc >> 4, r = bc & 15;
            int ln, j;
            if      (pl <  8) { ln = r;      j = pl;            }
            else if (pl < 16) { ln = 16 + r; j = pl - 8;        }
            else if (pl < 24) { ln = r;      j = 8 + (pl - 16); }
            else              { ln = 16 + r; j = 8 + (pl - 24); }
            u.s1.A[m][ln][j] = (_Float16)v;
            if (k0 + KT < PQ)   // warm L2/WGP$ for the next tile
                __builtin_prefetch(&data[((bc >> 3) * PQ + (k0 + KT + pl)) * CIN + (bc & 7)], 0, 1);
        }

        // ---- synthesize filter tile into B-fragment layout (16-bit B 32x16) ----
        for (int i = t; i < KT * QT; i += NTHR) {
            int pl = i >> 4;                 // local p
            int ql = i & 15;                 // local q
            int p  = k0 + pl;
            float dy  = grid[p]      - sQy[ql];
            float dx  = grid[PQ + p] - sQx[ql];
            float rad = sqrtf(dy * dy + dx * dx);
            // cos/sin of theta = atan2(dy,dx) without atan2: dx/r, dy/r.
            // rad==0 -> rinv=0 keeps everything finite; masked below for f!=0.
            float rinv = (rad > 0.0f) ? __builtin_amdgcn_rcpf(rad) : 0.0f;
            float c1 = dx * rinv;
            float s1 = dy * rinv;
            float c2 = __builtin_fmaf(2.0f * c1, c1, -1.0f);   // cos 2t
            float s2 = 2.0f * s1 * c1;                          // sin 2t
            float c3 = c2 * c1 - s2 * s1;                       // cos 3t
            float s3 = s2 * c1 + c2 * s1;                       // sin 3t
            int   ln  = ql + (((pl >> 3) & 1) ? 16 : 0);
            int   j   = (pl & 7) + ((pl >> 4) << 3);
#pragma unroll
            for (int f = 0; f < NF; ++f) {
                float d = (rad - br[f]) * binv[f];
                float e = __expf(-d * d);                       // Gaussian ring envelope
                int   a = anf[f];                               // uniform per f
                float cc = (a == 0) ? 1.0f : (a == 1) ? c1 : (a == 2) ? c2 : c3;
                float ss = (a == 0) ? 0.0f : (a == 1) ? s1 : (a == 2) ? s2 : s3;
                ss *= fsgn[f];
                float keep = (rad != 0.0f || a == 0) ? 1.0f : 0.0f;
                e *= keep;
                u.s1.B[f][0][ln][j] = (_Float16)(e * cc);
                u.s1.B[f][1][ln][j] = (_Float16)(e * ss);
            }
        }
        __syncthreads();

        // ---- 4 WMMAs per wave: D = A x B + C, f32 accumulate ----
        v16h a0  = *(const v16h*)&u.s1.A[0][lane][0];
        v16h a1  = *(const v16h*)&u.s1.A[1][lane][0];
        v16h bre = *(const v16h*)&u.s1.B[w][0][lane][0];
        v16h bim = *(const v16h*)&u.s1.B[w][1][lane][0];
        acc00 = __builtin_amdgcn_wmma_f32_16x16x32_f16(false, a0, false, bre, (short)0, acc00, false, false);
        acc10 = __builtin_amdgcn_wmma_f32_16x16x32_f16(false, a1, false, bre, (short)0, acc10, false, false);
        acc01 = __builtin_amdgcn_wmma_f32_16x16x32_f16(false, a0, false, bim, (short)0, acc01, false, false);
        acc11 = __builtin_amdgcn_wmma_f32_16x16x32_f16(false, a1, false, bim, (short)0, acc11, false, false);
    }

    // ---- spill accumulators to LDS per documented C/D layout ----
    {
        int ql = lane & 15;
        int hi = lane >> 4;                  // 0 or 1
#pragma unroll
        for (int i = 0; i < 8; ++i) {
            int M   = i + 8 * hi;
            int bc0 = M;                     // bc-tile 0
            int bc1 = 16 + M;                // bc-tile 1
            sPP[bc0 >> 3][ql][bc0 & 7][w][0] = acc00[i];
            sPP[bc0 >> 3][ql][bc0 & 7][w][1] = acc01[i];
            sPP[bc1 >> 3][ql][bc1 & 7][w][0] = acc10[i];
            sPP[bc1 >> 3][ql][bc1 & 7][w][1] = acc11[i];
        }
    }
    __syncthreads();   // fragments dead; safe to overlay u.s2

    // ---- build dense wfull + premul tables in the overlaid region ----
    for (int i = t; i < FSEL * COUT * 2; i += NTHR) ((float*)u.s2.W)[i] = 0.0f;
    if (t < FSEL) {
        u.s2.Idx[t]    = pre_cut_idx[t];
        u.s2.Mul[t][0] = pre_mul[2 * t];
        u.s2.Mul[t][1] = pre_mul[2 * t + 1];
    }
    __syncthreads();
    for (int n = t; n < nnz; n += NTHR) {
        int r = conn_rows[n], c = conn_cols[n];
        u.s2.W[r][c][0] = weights[2 * n];
        u.s2.W[r][c][1] = weights[2 * n + 1];
    }
    __syncthreads();

    // ---- second (tiny) complex GEMM: [NB*QT, FSEL] x [FSEL, COUT] ----
    for (int o = t; o < NB * QT * COUT; o += NTHR) {   // exactly 8 per thread
        int co  = o % COUT;
        int rem = o / COUT;
        int ql  = rem % QT;
        int b   = rem / QT;
        float are = 0.0f, aim = 0.0f;
#pragma unroll 4
        for (int fs = 0; fs < FSEL; ++fs) {
            int idx = u.s2.Idx[fs];
            int c   = idx / NF;
            int f   = idx - c * NF;
            float pr = sPP[b][ql][c][f][0] * u.s2.Mul[fs][0];
            float pi = sPP[b][ql][c][f][1] * u.s2.Mul[fs][1];
            float wr = u.s2.W[fs][co][0];
            float wi = u.s2.W[fs][co][1];
            are += pr * wr - pi * wi;
            aim += pr * wi + pi * wr;
        }
        if (out_rot[co] == 0) aim = 0.0f;    // keep_imag mask
        float* op = out + (((size_t)b * PQ + (q0 + ql)) * COUT + co) * 2;
        op[0] = are;
        op[1] = aim;
    }
}

extern "C" void kernel_launch(void* const* d_in, const int* in_sizes, int n_in,
                              void* d_out, int out_size, void* d_ws, size_t ws_size,
                              hipStream_t stream) {
    const float* data        = (const float*)d_in[0];
    const float* grid        = (const float*)d_in[1];
    const float* weights     = (const float*)d_in[2];
    const float* b_rad       = (const float*)d_in[3];
    const float* b_sig       = (const float*)d_in[4];
    const float* b_frq       = (const float*)d_in[5];
    const float* pre_mul     = (const float*)d_in[6];
    const int*   pre_cut_idx = (const int*)d_in[7];
    const int*   conn_rows   = (const int*)d_in[8];
    const int*   conn_cols   = (const int*)d_in[9];
    const int*   out_rot     = (const int*)d_in[10];
    const int    nnz         = in_sizes[8];

    flexpconv_fused<<<dim3(PQ / QT), dim3(NTHR), 0, stream>>>(
        data, grid, weights, b_rad, b_sig, b_frq,
        pre_mul, pre_cut_idx, conn_rows, conn_cols, out_rot,
        (float*)d_out, nnz);
}
